// seRNN_13486197309667
// MI455X (gfx1250) — matrix operand order repaired and checked by reading, not compiled, not run on hardware
//
#include <hip/hip_runtime.h>
#include <hip/hip_bf16.h>
#include <math.h>

typedef __attribute__((ext_vector_type(16))) __bf16 v16bf;
typedef __attribute__((ext_vector_type(8)))  __bf16 v8bf;
typedef __attribute__((ext_vector_type(4)))  __bf16 v4bf;
typedef __attribute__((ext_vector_type(8)))  float  v8f;

#define B_  64
#define T_  512
#define D_  512
#define H_  1024

// LDS row strides (elements + pad) to spread banks.
#define P1_KP 520      // phase 1: W_in slice rows of 512 bf16 (+8 pad)
#define WPAD  1032     // phase 2: W_hh slice rows of 1024 bf16 (+8 pad)

static __device__ __forceinline__ v16bf cvt16(float4 a, float4 b, float4 c, float4 d) {
  v16bf r;
  r[0]  = (__bf16)a.x; r[1]  = (__bf16)a.y; r[2]  = (__bf16)a.z; r[3]  = (__bf16)a.w;
  r[4]  = (__bf16)b.x; r[5]  = (__bf16)b.y; r[6]  = (__bf16)b.z; r[7]  = (__bf16)b.w;
  r[8]  = (__bf16)c.x; r[9]  = (__bf16)c.y; r[10] = (__bf16)c.z; r[11] = (__bf16)c.w;
  r[12] = (__bf16)d.x; r[13] = (__bf16)d.y; r[14] = (__bf16)d.z; r[15] = (__bf16)d.w;
  return r;
}

// 16 contiguous bf16 -> one fragment (two adjacent 16B loads).
static __device__ __forceinline__ v16bf ld16(const __bf16* p) {
  const v8bf* q = (const v8bf*)p;
  return __builtin_shufflevector(q[0], q[1],
      0,1,2,3,4,5,6,7,8,9,10,11,12,13,14,15);
}

// A-fragment loader: two adjacent 8-element K-groups (p, p+8) = 16 contiguous K,
// matching the B fragment order with the lane-half offset folded into p.
static __device__ __forceinline__ v16bf ld16_pair(const __bf16* p) {
  const v8bf* lo = (const v8bf*)p;
  const v8bf* hi = (const v8bf*)(p + 8);
  return __builtin_shufflevector(lo[0], hi[0],
      0,1,2,3,4,5,6,7,8,9,10,11,12,13,14,15);
}

// ---------------------------------------------------------------------------
// Workspace layout:
//   [0, 256)           : 4 per-group barrier counters (unsigned), padded
//   [256, 256 + 256KB) : h double buffer: [2 bufs][4 groups][16][H] bf16
// ---------------------------------------------------------------------------
#define WS_HOFF   256
#define WS_WORDS  ((WS_HOFF + 2 * 4 * 16 * H_ * 2) / 4)   // 65600 dwords

__global__ void rnn_ws_init(unsigned* __restrict__ ws_u, int nwords) {
  int i = blockIdx.x * blockDim.x + threadIdx.x;
  if (i < nwords) ws_u[i] = 0u;
}

// ---------------------------------------------------------------------------
// Phase 1: xin = x @ W_in^T + b_in + bias   (M=32768, N=1024, K=512) -> d_out
// Block: 256 thr = 8 waves; block tile M=128 x N=128.
// W_in slice (128 x 512) converted once into LDS (bf16); each wave computes
// a 16(M) x 128(N) strip: 8 WMMA accumulators, A fragment reused 8x.
// B fragments preloaded into a register array so the ds_loads clause up and
// WMMAs retire against staggered dscnt waits.
// ---------------------------------------------------------------------------
__global__ __launch_bounds__(256) void rnn_xin_gemm(
    const float* __restrict__ x, const float* __restrict__ W_in,
    const float* __restrict__ b_in, const float* __restrict__ bias,
    float* __restrict__ out) {
  extern __shared__ char smem[];
  __bf16* b_lds = (__bf16*)smem;                  // [128][P1_KP]

  const int tid  = threadIdx.x;
  const int Nblk = blockIdx.y * 128;

  // Stage + convert W_in slice: 128 rows x 512 K, float4 chunks.
  for (int idx = tid; idx < 128 * 128; idx += 256) {
    const int row = idx >> 7;
    const int c4  = idx & 127;
    float4 wv = *(const float4*)(W_in + (size_t)(Nblk + row) * D_ + c4 * 4);
    v4bf pk = { (__bf16)wv.x, (__bf16)wv.y, (__bf16)wv.z, (__bf16)wv.w };
    *(v4bf*)(b_lds + row * P1_KP + c4 * 4) = pk;
  }
  __syncthreads();

  const int lane  = tid & 31;
  const int wid   = tid >> 5;
  const int lmod  = lane & 15;
  const int lhalf = lane >> 4;
  const int Mbase = blockIdx.x * 128 + wid * 16;

  v8f acc[8] = {v8f{}, v8f{}, v8f{}, v8f{}, v8f{}, v8f{}, v8f{}, v8f{}};
  const float* xrow = x + (size_t)(Mbase + lmod) * D_;
  const __bf16* bbase = b_lds + (size_t)lmod * P1_KP + 16 * lhalf;

#pragma unroll 2
  for (int kb = 0; kb < D_; kb += 32) {
    // A fragment: lane holds row M=lmod, 16 contiguous K [kb+16*lhalf,+16).
    const float* ap = xrow + kb + 16 * lhalf;
    float4 a0 = *(const float4*)(ap + 0);
    float4 a1 = *(const float4*)(ap + 4);
    float4 a2 = *(const float4*)(ap + 8);
    float4 a3 = *(const float4*)(ap + 12);
    v16bf A = cvt16(a0, a1, a2, a3);

    // Preload all 8 B fragments (lane = col N, 16 contiguous K).
    v16bf Bfr[8];
#pragma unroll
    for (int nt = 0; nt < 8; ++nt)
      Bfr[nt] = ld16(bbase + (size_t)nt * (16 * P1_KP) + kb);

#pragma unroll
    for (int nt = 0; nt < 8; ++nt)
      acc[nt] = __builtin_amdgcn_wmma_f32_16x16x32_bf16(
          false, A, false, Bfr[nt], (short)0, acc[nt], false, false);
  }

#pragma unroll
  for (int nt = 0; nt < 8; ++nt) {
    const int n = Nblk + nt * 16 + lmod;
    const float bn = b_in[n] + bias[n];
#pragma unroll
    for (int v = 0; v < 8; ++v) {
      const int m = Mbase + v + 8 * lhalf;
      out[(size_t)m * H_ + n] = acc[nt][v] + bn;
    }
  }
}

// ---------------------------------------------------------------------------
// Phase 2: persistent scan. 64 WGs x 128 thr (4 waves).
//   group g = blockIdx.x/16 : batch rows [16g, 16g+16)
//   wg      = blockIdx.x%16 : output cols [64*wg, 64*wg+64)
// W_hh slice (64 x 1024) held in LDS as bf16. h state double-buffered in d_ws
// (bf16, L2-resident). Per-step sync: atomic arrive counter per group.
// xin loads hoisted ahead of the GEMM; two independent WMMA chains.
// ---------------------------------------------------------------------------
__global__ __launch_bounds__(128) void rnn_scan(
    const float* __restrict__ W_hh, float* __restrict__ out,
    void* __restrict__ ws) {
  extern __shared__ char smem[];
  __bf16* w_lds = (__bf16*)smem;                  // [64][WPAD]

  const int g     = blockIdx.x >> 4;
  const int wg    = blockIdx.x & 15;
  const int nbase = wg * 64;
  const int tid   = threadIdx.x;

  // Preload + convert W_hh slice into LDS (row n_local, 1024 K each).
  for (int idx = tid; idx < 64 * 256; idx += 128) {
    const int row = idx >> 8;
    const int c4  = idx & 255;
    float4 wv = *(const float4*)(W_hh + (size_t)(nbase + row) * H_ + c4 * 4);
    v4bf pk = { (__bf16)wv.x, (__bf16)wv.y, (__bf16)wv.z, (__bf16)wv.w };
    *(v4bf*)(w_lds + row * WPAD + c4 * 4) = pk;
  }
  __syncthreads();

  unsigned* cnt  = (unsigned*)ws + g;
  __bf16* hbase  = (__bf16*)((char*)ws + WS_HOFF);

  const int lane  = tid & 31;
  const int wv_id = tid >> 5;                     // wave 0..3 -> N tile
  const int lmod  = lane & 15;
  const int lhalf = lane >> 4;
  const int nloc  = wv_id * 16 + lmod;
  const int n     = nbase + nloc;
  const __bf16* brow = w_lds + (size_t)nloc * WPAD + 16 * lhalf;

  for (int t = 0; t < T_; ++t) {
    const __bf16* hcur = hbase + (size_t)(((t & 1) * 4 + g)) * (16 * H_);
    __bf16*       hnxt = hbase + (size_t)((((t + 1) & 1) * 4 + g)) * (16 * H_);

    // Hoist xin loads (d_out holds phase-1 result): overlap with GEMM below.
    float xin[8];
#pragma unroll
    for (int v = 0; v < 8; ++v) {
      const int mloc = v + 8 * lhalf;
      const int m    = g * 16 + mloc;
      xin[v] = out[((size_t)m * T_ + t) * H_ + n];
    }

    v8f acc0 = {};
    v8f acc1 = {};
    const __bf16* arow = hcur + (size_t)lmod * H_ + 16 * lhalf;

#pragma unroll 4
    for (int kb = 0; kb < H_; kb += 64) {
      // Preload fragments for both chains, then issue both WMMAs.
      v16bf A0 = ld16_pair(arow + kb);
      v16bf A1 = ld16_pair(arow + kb + 32);
      v16bf B0 = ld16(brow + kb);
      v16bf B1 = ld16(brow + kb + 32);
      acc0 = __builtin_amdgcn_wmma_f32_16x16x32_bf16(
          false, A0, false, B0, (short)0, acc0, false, false);
      acc1 = __builtin_amdgcn_wmma_f32_16x16x32_bf16(
          false, A1, false, B1, (short)0, acc1, false, false);
    }

    // Epilogue: h_new = tanh(xin + h@W^T); write back to d_out + next h buf.
#pragma unroll
    for (int v = 0; v < 8; ++v) {
      const int mloc = v + 8 * lhalf;
      const int m    = g * 16 + mloc;
      const size_t oidx = ((size_t)m * T_ + t) * H_ + n;
      float val = tanhf(acc0[v] + acc1[v] + xin[v]);
      out[oidx] = val;
      hnxt[(size_t)mloc * H_ + n] = (__bf16)val;
    }

    __threadfence();
    __syncthreads();
    if (tid == 0) {
      const unsigned target = (unsigned)(t + 1) * 16u;
      __hip_atomic_fetch_add(cnt, 1u, __ATOMIC_RELEASE,
                             __HIP_MEMORY_SCOPE_AGENT);
      while (__hip_atomic_load(cnt, __ATOMIC_ACQUIRE,
                               __HIP_MEMORY_SCOPE_AGENT) < target)
        __builtin_amdgcn_s_sleep(2);
    }
    __syncthreads();
  }
}

// ---------------------------------------------------------------------------
extern "C" void kernel_launch(void* const* d_in, const int* in_sizes, int n_in,
                              void* d_out, int out_size, void* d_ws, size_t ws_size,
                              hipStream_t stream) {
  (void)in_sizes; (void)n_in; (void)out_size; (void)ws_size;
  const float* x    = (const float*)d_in[0];
  const float* W_in = (const float*)d_in[1];
  const float* b_in = (const float*)d_in[2];
  const float* W_hh = (const float*)d_in[3];
  const float* bias = (const float*)d_in[4];
  float* out = (float*)d_out;

  // 1) reset barrier counters + zero h0 double buffer (ws not re-poisoned).
  rnn_ws_init<<<(WS_WORDS + 255) / 256, 256, 0, stream>>>(
      (unsigned*)d_ws, WS_WORDS);

  // 2) xin = x @ W_in^T + b_in + bias  -> d_out  (W_in slice staged in LDS)
  dim3 g1((B_ * T_) / 128, H_ / 128);
  rnn_xin_gemm<<<g1, 256, 130 * 1024, stream>>>(x, W_in, b_in, bias, out);

  // 3) persistent recurrent scan (64 WGs, 129 KB dynamic LDS each).
  rnn_scan<<<64, 128, 64 * WPAD * 2, stream>>>(W_hh, out, d_ws);
}
